// Attention_36661840839167
// MI455X (gfx1250) — compile-verified, matrix operand order checked by reading, not run on hardware
//
#include <hip/hip_runtime.h>
#include <hip/hip_bf16.h>

// ---------------------------------------------------------------------------
// Multi-head attention on gfx1250 using v_wmma_f32_16x16x32_bf16.
// Pipeline: cvt(bf16) -> proj GEMMs -> flash attention (async-LDS staged,
// double buffered) -> output GEMM + bias.
// ---------------------------------------------------------------------------

typedef __attribute__((ext_vector_type(16))) __bf16 v16bf;
typedef __attribute__((ext_vector_type(8)))  float  v8f;
typedef __attribute__((ext_vector_type(4)))  int    v4i;

#define BATCH 4
#define HEADS 8
#define NQL   2048
#define NKL   2048
#define HD    64
#define INNER 512
#define DIMQ  512
#define DIMKV 768

#define KSTR  72   // LDS row stride (halves) for K tile: 144B, 16B-aligned rows
#define VSTR  40   // LDS row stride (halves) for V tile / P tile: 80B rows

union FragBF {
  v16bf v;
  uint4 q[2];
};

__device__ __forceinline__ unsigned short f2bf(float f) {
  unsigned int u = __float_as_uint(f);
  u += 0x7fffu + ((u >> 16) & 1u);   // round-to-nearest-even
  return (unsigned short)(u >> 16);
}

// Load one 16x32 bf16 fragment for this lane: row pointer p, lane k-half khalf.
// Element i of the fragment holds K = (i/8)*16 + khalf*8 + (i%8).
__device__ __forceinline__ void load_frag_g(FragBF& f, const unsigned short* p, int khalf) {
  f.q[0] = *(const uint4*)(p + khalf * 8);
  f.q[1] = *(const uint4*)(p + 16 + khalf * 8);
}

__device__ __forceinline__ v8f wmma_bf16(const FragBF& a, const FragBF& b, v8f c) {
  return __builtin_amdgcn_wmma_f32_16x16x32_bf16(
      /*neg_a=*/false, a.v, /*neg_b=*/false, b.v,
      /*c_mod=*/(short)0, c, /*reuse_a=*/false, /*reuse_b=*/false);
}

// ---------------------------------------------------------------------------
// gfx1250 async global->LDS copy (ASYNCcnt-tracked). Fallback: staged via regs.
// ---------------------------------------------------------------------------
#if __has_builtin(__builtin_amdgcn_global_load_async_to_lds_b128)
#define HAVE_ASYNC_LDS 1
#else
#define HAVE_ASYNC_LDS 0
#endif

__device__ __forceinline__ void async_cp16(const unsigned short* g, unsigned short* l) {
#if HAVE_ASYNC_LDS
  __builtin_amdgcn_global_load_async_to_lds_b128(
      (__attribute__((address_space(1))) v4i*)g,
      (__attribute__((address_space(3))) v4i*)l, 0, 0);
#else
  *(uint4*)l = *(const uint4*)g;
#endif
}

#if __has_builtin(__builtin_amdgcn_s_wait_asynccnt)
#define WAIT_ASYNC(n) __builtin_amdgcn_s_wait_asynccnt(n)
#elif HAVE_ASYNC_LDS
#define WAIT_ASYNC(n) __asm__ volatile("s_wait_asynccnt %0" ::"i"(n) : "memory")
#else
#define WAIT_ASYNC(n) ((void)0)
#endif

// ---------------------------------------------------------------------------
// Elementwise f32 -> bf16
// ---------------------------------------------------------------------------
__global__ void cvt_bf16_kernel(const float* __restrict__ in,
                                unsigned short* __restrict__ out, int n) {
  int i = blockIdx.x * blockDim.x + threadIdx.x;
  if (i < n) out[i] = f2bf(in[i]);
}

// W[K,N] f32 -> Wt[N,K] bf16 (transpose so B-operand columns become rows)
__global__ void cvt_tr_kernel(const float* __restrict__ W,
                              unsigned short* __restrict__ Wt, int K, int N) {
  int i = blockIdx.x * blockDim.x + threadIdx.x;
  if (i >= K * N) return;
  int k = i / N, n = i % N;
  Wt[(size_t)n * K + k] = f2bf(W[i]);
}

// ---------------------------------------------------------------------------
// Projection GEMM: C[M,Nn] = A[M,K] @ Bt[Nn,K]^T, bf16 in, bf16 out.
// Block = 8 waves; wave w handles rows [blockIdx.x*128 + 16w, +16), 64 cols
// (= one head, h = blockIdx.y). mode 0: out [B,H,N,64]; mode 1: out [B,H,64,N].
// ---------------------------------------------------------------------------
__global__ __launch_bounds__(256) void proj_gemm_kernel(
    const unsigned short* __restrict__ A,
    const unsigned short* __restrict__ Bt,
    unsigned short* __restrict__ out,
    int K, int mode, float scale) {
  const int lane  = threadIdx.x & 31;
  const int wave  = threadIdx.x >> 5;
  const int l16   = lane & 15;
  const int khalf = lane >> 4;
  const int mbase = blockIdx.x * 128 + wave * 16;
  const int h     = blockIdx.y;

  const unsigned short* arow = A + (size_t)(mbase + l16) * K;
  const unsigned short* b0 = Bt + (size_t)(h * 64 + 0 * 16 + l16) * K;
  const unsigned short* b1 = Bt + (size_t)(h * 64 + 1 * 16 + l16) * K;
  const unsigned short* b2 = Bt + (size_t)(h * 64 + 2 * 16 + l16) * K;
  const unsigned short* b3 = Bt + (size_t)(h * 64 + 3 * 16 + l16) * K;

  v8f c0 = {}, c1 = {}, c2 = {}, c3 = {};
  for (int k = 0; k < K; k += 32) {
    FragBF a, bb;
    load_frag_g(a, arow + k, khalf);
    load_frag_g(bb, b0 + k, khalf); c0 = wmma_bf16(a, bb, c0);
    load_frag_g(bb, b1 + k, khalf); c1 = wmma_bf16(a, bb, c1);
    load_frag_g(bb, b2 + k, khalf); c2 = wmma_bf16(a, bb, c2);
    load_frag_g(bb, b3 + k, khalf); c3 = wmma_bf16(a, bb, c3);
  }

  const int batch = mbase / NQL;
  const int rbase = mbase % NQL;
#pragma unroll
  for (int r = 0; r < 8; ++r) {
    const int rr = rbase + khalf * 8 + r;
    if (mode == 0) {
      unsigned short* o = out + (((size_t)batch * HEADS + h) * NQL + rr) * HD;
      o[0 * 16 + l16] = f2bf(c0[r] * scale);
      o[1 * 16 + l16] = f2bf(c1[r] * scale);
      o[2 * 16 + l16] = f2bf(c2[r] * scale);
      o[3 * 16 + l16] = f2bf(c3[r] * scale);
    } else {
      unsigned short* o = out + (((size_t)batch * HEADS + h) * HD) * (size_t)NKL + rr;
      o[(size_t)(0 * 16 + l16) * NKL] = f2bf(c0[r] * scale);
      o[(size_t)(1 * 16 + l16) * NKL] = f2bf(c1[r] * scale);
      o[(size_t)(2 * 16 + l16) * NKL] = f2bf(c2[r] * scale);
      o[(size_t)(3 * 16 + l16) * NKL] = f2bf(c3[r] * scale);
    }
  }
}

// ---------------------------------------------------------------------------
// Flash attention. Qh,Kh: [B*H, N, 64] bf16 (Q pre-scaled by 1/sqrt(64)).
// Vt: [B*H, 64, N] bf16. Aout: [B*N, 512] bf16 (cols = h*64+d).
// Each block: one (b,h), 128 query rows (8 waves x 16 rows).
// K/V tiles (32 keys) are staged into LDS once per block via async loads,
// double buffered; all waves read WMMA fragments from LDS.
// ---------------------------------------------------------------------------
__global__ __launch_bounds__(256) void attn_kernel(
    const unsigned short* __restrict__ Qh,
    const unsigned short* __restrict__ Kh,
    const unsigned short* __restrict__ Vt,
    unsigned short* __restrict__ Aout) {
  const int tid   = threadIdx.x;
  const int lane  = tid & 31;
  const int wave  = tid >> 5;
  const int l16   = lane & 15;
  const int khalf = lane >> 4;

  const int qtiles = NQL / 128;
  const int bh   = blockIdx.x / qtiles;
  const int qblk = blockIdx.x % qtiles;

  const unsigned short* qrow =
      Qh + ((size_t)bh * NQL + qblk * 128 + wave * 16 + l16) * HD;
  FragBF qa0, qa1;
  load_frag_g(qa0, qrow, khalf);        // d = 0..31
  load_frag_g(qa1, qrow + 32, khalf);   // d = 32..63

  v8f o0 = {}, o1 = {}, o2 = {}, o3 = {};
  float rm[8], rl[8];
#pragma unroll
  for (int r = 0; r < 8; ++r) { rm[r] = -1e30f; rl[r] = 0.0f; }

  // Double-buffered K (32 x 64) and V (64 x 32) tiles + per-wave P tile.
  __shared__ __align__(16) unsigned short kbuf[2][32 * KSTR];  // 9.0 KB
  __shared__ __align__(16) unsigned short vbuf[2][64 * VSTR];  // 10.0 KB
  __shared__ __align__(16) unsigned short pbuf[8][16 * VSTR];  // 10.0 KB
  unsigned short* pw = pbuf[wave];

  const unsigned short* kbase = Kh + (size_t)bh * NKL * HD;
  const unsigned short* vbase = Vt + (size_t)bh * HD * NKL;

  // 256 threads cover each 4KB tile with one b128 chunk per thread.
  const int krow = tid >> 3, kch = tid & 7;   // 32 rows x 8 chunks of 16B
  const int vrow = tid >> 2, vch = tid & 3;   // 64 rows x 4 chunks of 16B

  auto issue_stage = [&](int s, int buf) {
    async_cp16(kbase + (size_t)(s * 32 + krow) * HD + kch * 8,
               &kbuf[buf][krow * KSTR + kch * 8]);
    async_cp16(vbase + (size_t)vrow * NKL + s * 32 + vch * 8,
               &vbuf[buf][vrow * VSTR + vch * 8]);
  };

  const int nst = NKL / 32;   // 64 stages
  issue_stage(0, 0);

  for (int s = 0; s < nst; ++s) {
    const int cur = s & 1;
    if (s + 1 < nst) {
      issue_stage(s + 1, cur ^ 1);   // prefetch next tile into other buffer
      WAIT_ASYNC(2);                 // this wave's stage-s copies complete
    } else {
      WAIT_ASYNC(0);
    }
    __syncthreads();                 // publish stage-s tiles to all waves

    const unsigned short* kb = kbuf[cur];
    const unsigned short* vb = vbuf[cur];

    // ---- scores: two 16x16 tiles over keys s*32 .. s*32+31 ----
    FragBF f;
    v8f s0 = {}, s1 = {};
    load_frag_g(f, kb + l16 * KSTR, khalf);             s0 = wmma_bf16(qa0, f, s0);
    load_frag_g(f, kb + l16 * KSTR + 32, khalf);        s0 = wmma_bf16(qa1, f, s0);
    load_frag_g(f, kb + (16 + l16) * KSTR, khalf);      s1 = wmma_bf16(qa0, f, s1);
    load_frag_g(f, kb + (16 + l16) * KSTR + 32, khalf); s1 = wmma_bf16(qa1, f, s1);

    // ---- online softmax update (row = khalf*8 + r, cols across 16 lanes) ----
#pragma unroll
    for (int r = 0; r < 8; ++r) {
      float tm = fmaxf(s0[r], s1[r]);
      tm = fmaxf(tm, __shfl_xor(tm, 1, 32));
      tm = fmaxf(tm, __shfl_xor(tm, 2, 32));
      tm = fmaxf(tm, __shfl_xor(tm, 4, 32));
      tm = fmaxf(tm, __shfl_xor(tm, 8, 32));
      const float mnew  = fmaxf(rm[r], tm);
      const float alpha = __expf(rm[r] - mnew);
      const float p0 = __expf(s0[r] - mnew);
      const float p1 = __expf(s1[r] - mnew);
      float rs = p0 + p1;
      rs += __shfl_xor(rs, 1, 32);
      rs += __shfl_xor(rs, 2, 32);
      rs += __shfl_xor(rs, 4, 32);
      rs += __shfl_xor(rs, 8, 32);
      rl[r] = rl[r] * alpha + rs;
      rm[r] = mnew;
      o0[r] *= alpha; o1[r] *= alpha; o2[r] *= alpha; o3[r] *= alpha;
      const int row = khalf * 8 + r;
      pw[row * VSTR + l16]      = f2bf(p0);
      pw[row * VSTR + 16 + l16] = f2bf(p1);
    }
    __asm__ volatile("s_wait_dscnt 0" ::: "memory");  // wave-local LDS RAW

    // reload P as a 16x32 A fragment (C-layout -> A-layout via LDS)
    FragBF pa;
    pa.q[0] = *(const uint4*)(pw + l16 * VSTR + khalf * 8);
    pa.q[1] = *(const uint4*)(pw + l16 * VSTR + 16 + khalf * 8);

    // ---- O += P @ V : 4 d-tiles, B-operand rows from LDS V tile ----
    load_frag_g(f, vb + (0 * 16 + l16) * VSTR, khalf); o0 = wmma_bf16(pa, f, o0);
    load_frag_g(f, vb + (1 * 16 + l16) * VSTR, khalf); o1 = wmma_bf16(pa, f, o1);
    load_frag_g(f, vb + (2 * 16 + l16) * VSTR, khalf); o2 = wmma_bf16(pa, f, o2);
    load_frag_g(f, vb + (3 * 16 + l16) * VSTR, khalf); o3 = wmma_bf16(pa, f, o3);

    __syncthreads();   // all waves done with stage-s tiles before refill
  }

  // ---- normalize, write Aout[b*N + q, h*64 + d] ----
  const int b = bh >> 3, h = bh & 7;
#pragma unroll
  for (int r = 0; r < 8; ++r) {
    const float inv = 1.0f / rl[r];
    const int m = khalf * 8 + r;
    const size_t row = (size_t)b * NQL + qblk * 128 + wave * 16 + m;
    unsigned short* orow = Aout + row * INNER + h * HD;
    orow[0 * 16 + l16] = f2bf(o0[r] * inv);
    orow[1 * 16 + l16] = f2bf(o1[r] * inv);
    orow[2 * 16 + l16] = f2bf(o2[r] * inv);
    orow[3 * 16 + l16] = f2bf(o3[r] * inv);
  }
}

// ---------------------------------------------------------------------------
// Output GEMM: out[M,512] f32 = A[M,512]bf16 @ WoT[512,512]^T + bo
// ---------------------------------------------------------------------------
__global__ __launch_bounds__(256) void out_gemm_kernel(
    const unsigned short* __restrict__ A,
    const unsigned short* __restrict__ Bt,
    const float* __restrict__ bias,
    float* __restrict__ out, int K) {
  const int lane  = threadIdx.x & 31;
  const int wave  = threadIdx.x >> 5;
  const int l16   = lane & 15;
  const int khalf = lane >> 4;
  const int mbase = blockIdx.x * 128 + wave * 16;
  const int nblk  = blockIdx.y;

  const unsigned short* arow = A + (size_t)(mbase + l16) * K;
  const unsigned short* b0 = Bt + (size_t)(nblk * 64 + 0 * 16 + l16) * K;
  const unsigned short* b1 = Bt + (size_t)(nblk * 64 + 1 * 16 + l16) * K;
  const unsigned short* b2 = Bt + (size_t)(nblk * 64 + 2 * 16 + l16) * K;
  const unsigned short* b3 = Bt + (size_t)(nblk * 64 + 3 * 16 + l16) * K;

  v8f c0 = {}, c1 = {}, c2 = {}, c3 = {};
  for (int k = 0; k < K; k += 32) {
    FragBF a, bb;
    load_frag_g(a, arow + k, khalf);
    load_frag_g(bb, b0 + k, khalf); c0 = wmma_bf16(a, bb, c0);
    load_frag_g(bb, b1 + k, khalf); c1 = wmma_bf16(a, bb, c1);
    load_frag_g(bb, b2 + k, khalf); c2 = wmma_bf16(a, bb, c2);
    load_frag_g(bb, b3 + k, khalf); c3 = wmma_bf16(a, bb, c3);
  }

  const float bv0 = bias[nblk * 64 + 0 * 16 + l16];
  const float bv1 = bias[nblk * 64 + 1 * 16 + l16];
  const float bv2 = bias[nblk * 64 + 2 * 16 + l16];
  const float bv3 = bias[nblk * 64 + 3 * 16 + l16];
#pragma unroll
  for (int r = 0; r < 8; ++r) {
    const size_t row = (size_t)mbase + khalf * 8 + r;
    float* o = out + row * DIMQ + nblk * 64;
    o[0 * 16 + l16] = c0[r] + bv0;
    o[1 * 16 + l16] = c1[r] + bv1;
    o[2 * 16 + l16] = c2[r] + bv2;
    o[3 * 16 + l16] = c3[r] + bv3;
  }
}

// ---------------------------------------------------------------------------
// Host launcher
// ---------------------------------------------------------------------------
extern "C" void kernel_launch(void* const* d_in, const int* in_sizes, int n_in,
                              void* d_out, int out_size, void* d_ws, size_t ws_size,
                              hipStream_t stream) {
  (void)in_sizes; (void)n_in; (void)out_size; (void)ws_size;
  const float* q  = (const float*)d_in[0];
  const float* kv = (const float*)d_in[1];
  const float* Wq = (const float*)d_in[2];
  const float* Wk = (const float*)d_in[3];
  const float* Wv = (const float*)d_in[4];
  const float* Wo = (const float*)d_in[5];
  const float* bo = (const float*)d_in[6];
  float* out = (float*)d_out;

  char* ws = (char*)d_ws;
  size_t off = 0;
  auto alloc = [&](size_t bytes) -> void* {
    void* p = ws + off;
    off += (bytes + 255) & ~(size_t)255;
    return p;
  };

  const int M = BATCH * NQL;                       // 8192 rows for q and kv
  unsigned short* qb   = (unsigned short*)alloc((size_t)M * DIMQ * 2);
  unsigned short* kvb  = (unsigned short*)alloc((size_t)M * DIMKV * 2);
  unsigned short* WqT  = (unsigned short*)alloc((size_t)INNER * DIMQ * 2);
  unsigned short* WkT  = (unsigned short*)alloc((size_t)INNER * DIMKV * 2);
  unsigned short* WvT  = (unsigned short*)alloc((size_t)INNER * DIMKV * 2);
  unsigned short* WoT  = (unsigned short*)alloc((size_t)DIMQ * INNER * 2);
  unsigned short* Qh   = (unsigned short*)alloc((size_t)M * INNER * 2);
  unsigned short* Kh   = (unsigned short*)alloc((size_t)M * INNER * 2);
  unsigned short* Vt   = (unsigned short*)alloc((size_t)M * INNER * 2);
  unsigned short* Aout = (unsigned short*)alloc((size_t)M * INNER * 2);

  const int nq  = M * DIMQ;    // 4,194,304
  const int nkv = M * DIMKV;   // 6,291,456
  cvt_bf16_kernel<<<(nq + 255) / 256, 256, 0, stream>>>(q, qb, nq);
  cvt_bf16_kernel<<<(nkv + 255) / 256, 256, 0, stream>>>(kv, kvb, nkv);
  cvt_tr_kernel<<<(DIMQ * INNER + 255) / 256, 256, 0, stream>>>(Wq, WqT, DIMQ, INNER);
  cvt_tr_kernel<<<(DIMKV * INNER + 255) / 256, 256, 0, stream>>>(Wk, WkT, DIMKV, INNER);
  cvt_tr_kernel<<<(DIMKV * INNER + 255) / 256, 256, 0, stream>>>(Wv, WvT, DIMKV, INNER);
  cvt_tr_kernel<<<(INNER * DIMQ + 255) / 256, 256, 0, stream>>>(Wo, WoT, INNER, DIMQ);

  dim3 gp(M / 128, INNER / 64);   // 64 x 8 blocks, 256 threads (8 waves)
  const float qscale = 0.125f;    // 1/sqrt(HEAD_DIM), folded into Qh
  proj_gemm_kernel<<<gp, 256, 0, stream>>>(qb,  WqT, Qh, DIMQ,  0, qscale);
  proj_gemm_kernel<<<gp, 256, 0, stream>>>(kvb, WkT, Kh, DIMKV, 0, 1.0f);
  proj_gemm_kernel<<<gp, 256, 0, stream>>>(kvb, WvT, Vt, DIMKV, 1, 1.0f);

  attn_kernel<<<BATCH * HEADS * (NQL / 128), 256, 0, stream>>>(Qh, Kh, Vt, Aout);

  dim3 go(M / 128, DIMQ / 64);
  out_gemm_kernel<<<go, 256, 0, stream>>>(Aout, WoT, bo, out, INNER);
}
